// SelfAttention_34299608825909
// MI455X (gfx1250) — matrix operand order repaired and checked by reading, not compiled, not run on hardware
//
#include <hip/hip_runtime.h>
#include <hip/hip_bf16.h>

typedef __attribute__((ext_vector_type(16))) __bf16 v16bf;
typedef __attribute__((ext_vector_type(8)))  float  v8f;

#define SEQ    2048
#define EMBED  2048
#define HEADS  16
#define HDIM   128
#define NBATCH 2

#define SCALE_F  45.25483399593904f        /* sqrt(2048): ref scales energy UP */
#define MASKED_F (-4.5254833995939046e21f) /* -1e20 * sqrt(2048) */

// ---- POD helper vectors (HIP's float4/int4 have ctors -> unusable in unions)
struct alignas(16) U4 { unsigned x, y, z, w; };
struct alignas(16) F4 { float    x, y, z, w; };
struct alignas(16) I4 { int      x, y, z, w; };
union  Frag16 { U4 q[2]; unsigned u[8]; v16bf v; };

// round-to-nearest-even fp32 -> bf16 bits
__device__ __forceinline__ unsigned bfbits(float x) {
  unsigned u = __float_as_uint(x);
  return (u + 0x7fffu + ((u >> 16) & 1u)) >> 16;
}
__device__ __forceinline__ float bfval(float x) {
  return __uint_as_float(bfbits(x) << 16);
}
__device__ __forceinline__ unsigned pk2(float a, float b) {
  return bfbits(a) | (bfbits(b) << 16);
}

// =====================================================================
// Kernel 1: flash attention (transposed S/O formulation), software-pipelined.
//   block = 256 threads (8 waves); each wave: one 16-row q tile of (n,h).
//   Global loads for tile k+1 issue before tile k's compute (latency hidden
//   behind 32 WMMAs + softmax); LDS single-buffered.
//   Writes attention output [N, Q, H*D] fp32 to workspace.
// =====================================================================
__launch_bounds__(256, 1)
__global__ void attn_kernel(const float* __restrict__ Vg,
                            const float* __restrict__ Kg,
                            const float* __restrict__ Qg,
                            const int*   __restrict__ maskg,
                            float*       __restrict__ Og)
{
  const int bid  = blockIdx.x;
  const int qblk = bid & 15;          // SEQ/128 = 16
  const int h    = (bid >> 4) & 15;
  const int n    = bid >> 8;
  const int tid  = threadIdx.x;
  const int lane = tid & 31;
  const int wv   = tid >> 5;          // 0..7
  const int l16  = lane & 15;
  const int lh   = (lane >> 4) & 1;   // half-wave id
  const int q0   = qblk * 128 + wv * 16;

  // K tiles (hi/lo split) stored [key][dim], V tile stored transposed [dim][key].
  // Row paddings (+8 dims / +8 keys) make the 16-lane b128 fragment reads bank-conflict-free.
  __shared__ __align__(16) unsigned short Khi[32][136];
  __shared__ __align__(16) unsigned short Klo[32][136];
  __shared__ __align__(16) unsigned short Vt [128][40];

  const float* qb = Qg + (size_t)n * SEQ * EMBED + (size_t)h * HDIM;
  const float* kb = Kg + (size_t)n * SEQ * EMBED + (size_t)h * HDIM;
  const float* vb = Vg + (size_t)n * SEQ * EMBED + (size_t)h * HDIM;
  const int*   mb = maskg + n * SEQ;

  // ---- Q tile -> resident B-layout fragments (hi+lo), 4 chunks of K=32.
  // B layout (16-bit, 32x16): lane: N = q = l16, K = 16*lh + (0..15) contiguous.
  v16bf bqh[4], bql[4];
  {
    const float* qrow = qb + (size_t)(q0 + l16) * EMBED;
    #pragma unroll
    for (int c = 0; c < 4; ++c) {
      Frag16 hi, lo;
      const F4* p = (const F4*)(qrow + 32 * c + 16 * lh);
      #pragma unroll
      for (int j = 0; j < 4; ++j) {
        F4 f = p[j];
        float x0 = bfval(f.x), x1 = bfval(f.y), x2 = bfval(f.z), x3 = bfval(f.w);
        hi.u[2*j]   = pk2(f.x, f.y);
        hi.u[2*j+1] = pk2(f.z, f.w);
        lo.u[2*j]   = pk2(f.x - x0, f.y - x1);
        lo.u[2*j+1] = pk2(f.z - x2, f.w - x3);
      }
      bqh[c] = hi.v; bql[c] = lo.v;
    }
  }

  // cooperative-load geometry: 1024 f32x4 over 256 threads -> 4 per thread
  int ckey[4], cd4[4];
  #pragma unroll
  for (int i = 0; i < 4; ++i) {
    int idx = tid + i * 256;
    ckey[i] = idx >> 5;
    cd4[i]  = (idx & 31) << 2;
  }

  // ---- prologue: stage tile 0 in registers
  F4 kreg[4], vreg[4];
  #pragma unroll
  for (int i = 0; i < 4; ++i) {
    kreg[i] = *(const F4*)(kb + (size_t)ckey[i] * EMBED + cd4[i]);
    vreg[i] = *(const F4*)(vb + (size_t)ckey[i] * EMBED + cd4[i]);
  }

  float m_run = -__builtin_inff();
  float l_run = 0.0f;
  v8f o[8];                           // O^T accumulators: 8 d-tiles of 16x16
  #pragma unroll
  for (int t = 0; t < 8; ++t) {
    #pragma unroll
    for (int e = 0; e < 8; ++e) o[t][e] = 0.0f;
  }

  for (int k0 = 0; k0 < SEQ; k0 += 32) {
    // ---- commit staged tile to LDS (convert fp32 -> split bf16)
    __syncthreads();                  // all waves done reading previous tile
    #pragma unroll
    for (int i = 0; i < 4; ++i) {
      F4 kf = kreg[i], vf = vreg[i];
      int key = ckey[i], d4 = cd4[i];
      float x0 = bfval(kf.x), x1 = bfval(kf.y), x2 = bfval(kf.z), x3 = bfval(kf.w);
      unsigned* ph = (unsigned*)&Khi[key][d4];
      ph[0] = pk2(kf.x, kf.y); ph[1] = pk2(kf.z, kf.w);
      unsigned* pl = (unsigned*)&Klo[key][d4];
      pl[0] = pk2(kf.x - x0, kf.y - x1); pl[1] = pk2(kf.z - x2, kf.w - x3);
      Vt[d4+0][key] = (unsigned short)bfbits(vf.x);
      Vt[d4+1][key] = (unsigned short)bfbits(vf.y);
      Vt[d4+2][key] = (unsigned short)bfbits(vf.z);
      Vt[d4+3][key] = (unsigned short)bfbits(vf.w);
    }
    __syncthreads();

    // ---- issue global loads for next tile NOW; compute below hides latency
    if (k0 + 32 < SEQ) {
      #pragma unroll
      for (int i = 0; i < 4; ++i) {
        kreg[i] = *(const F4*)(kb + (size_t)(k0 + 32 + ckey[i]) * EMBED + cd4[i]);
        vreg[i] = *(const F4*)(vb + (size_t)(k0 + 32 + ckey[i]) * EMBED + cd4[i]);
      }
    }

    // ---- S^T = K_tile @ Q^T  (two 16x16 tiles; split-bf16: 3 WMMA per chunk)
    // A layout (16-bit, 16x32): lane: M = key = l16, K = 16*lh + (0..15) contiguous
    v8f s[2];
    #pragma unroll
    for (int sub = 0; sub < 2; ++sub) {
      v8f acc;
      #pragma unroll
      for (int e = 0; e < 8; ++e) acc[e] = 0.0f;
      #pragma unroll
      for (int c = 0; c < 4; ++c) {
        Frag16 ah, al;
        const unsigned short* rh = &Khi[sub*16 + l16][0];
        const unsigned short* rl = &Klo[sub*16 + l16][0];
        int o0 = 32*c + 16*lh, o1 = 32*c + 16*lh + 8;
        ah.q[0] = *(const U4*)(rh + o0); ah.q[1] = *(const U4*)(rh + o1);
        al.q[0] = *(const U4*)(rl + o0); al.q[1] = *(const U4*)(rl + o1);
        acc = __builtin_amdgcn_wmma_f32_16x16x32_bf16(false, ah.v, false, bqh[c], (short)0, acc, false, false);
        acc = __builtin_amdgcn_wmma_f32_16x16x32_bf16(false, ah.v, false, bql[c], (short)0, acc, false, false);
        acc = __builtin_amdgcn_wmma_f32_16x16x32_bf16(false, al.v, false, bqh[c], (short)0, acc, false, false);
      }
      s[sub] = acc;
    }

    // ---- mask (before scale, as in reference) + scale
    union { I4 v4[2]; int a[8]; } m0u, m1u;
    m0u.v4[0] = *(const I4*)(mb + k0 + 8*lh);
    m0u.v4[1] = *(const I4*)(mb + k0 + 8*lh + 4);
    m1u.v4[0] = *(const I4*)(mb + k0 + 16 + 8*lh);
    m1u.v4[1] = *(const I4*)(mb + k0 + 16 + 8*lh + 4);
    float sv0[8], sv1[8];
    float tm = -__builtin_inff();
    #pragma unroll
    for (int r = 0; r < 8; ++r) {
      sv0[r] = m0u.a[r] ? s[0][r] * SCALE_F : MASKED_F;
      sv1[r] = m1u.a[r] ? s[1][r] * SCALE_F : MASKED_F;
      tm = fmaxf(tm, fmaxf(sv0[r], sv1[r]));
    }
    // ---- online softmax; row (q) state lives per-lane, halves combined by one swap
    tm = fmaxf(tm, __shfl_xor(tm, 16, 32));
    float m_new = fmaxf(m_run, tm);
    float alpha = __expf(m_run - m_new);    // exp(-inf)=0 on first tile
    float p0[8], p1[8], ls = 0.0f;
    #pragma unroll
    for (int r = 0; r < 8; ++r) {
      p0[r] = __expf(sv0[r] - m_new);
      p1[r] = __expf(sv1[r] - m_new);
      ls += p0[r] + p1[r];
    }
    ls += __shfl_xor(ls, 16, 32);
    l_run = l_run * alpha + ls;
    m_run = m_new;
    #pragma unroll
    for (int t = 0; t < 8; ++t) {
      #pragma unroll
      for (int e = 0; e < 8; ++e) o[t][e] *= alpha;
    }

    // ---- P^T (C layout) -> B layout via half-wave swaps
    float oth0[8], oth1[8];
    #pragma unroll
    for (int r = 0; r < 8; ++r) {
      oth0[r] = __shfl_xor(p0[r], 16, 32);
      oth1[r] = __shfl_xor(p1[r], 16, 32);
    }
    Frag16 bp;
    #pragma unroll
    for (int j = 0; j < 4; ++j) {
      float a0 = lh ? oth1[2*j]   : p0[2*j];
      float a1 = lh ? oth1[2*j+1] : p0[2*j+1];
      float b0 = lh ? p1[2*j]     : oth0[2*j];
      float b1 = lh ? p1[2*j+1]   : oth0[2*j+1];
      bp.u[j]     = pk2(a0, a1);
      bp.u[j + 4] = pk2(b0, b1);
    }

    // ---- O^T += V^T @ P^T : 8 d-tiles, A from transposed V in LDS
    #pragma unroll
    for (int t = 0; t < 8; ++t) {
      Frag16 av;
      const unsigned short* vr = &Vt[16*t + l16][0];
      av.q[0] = *(const U4*)(vr + 16*lh);
      av.q[1] = *(const U4*)(vr + 16*lh + 8);
      o[t] = __builtin_amdgcn_wmma_f32_16x16x32_bf16(false, av.v, false, bp.v, (short)0, o[t], false, false);
    }
  }

  // ---- normalize and store: lane holds q = q0+l16, d contiguous along VGPRs
  float inv = 1.0f / l_run;
  float* orow = Og + (size_t)(n * SEQ + q0 + l16) * EMBED + h * HDIM + 8 * lh;
  #pragma unroll
  for (int t = 0; t < 8; ++t) {
    F4 f0, f1;
    f0.x = o[t][0]*inv; f0.y = o[t][1]*inv; f0.z = o[t][2]*inv; f0.w = o[t][3]*inv;
    f1.x = o[t][4]*inv; f1.y = o[t][5]*inv; f1.z = o[t][6]*inv; f1.w = o[t][7]*inv;
    *(F4*)(orow + 16*t)     = f0;
    *(F4*)(orow + 16*t + 4) = f1;
  }
}

// =====================================================================
// Kernel 2: out[i,j] = sum_k A[i,k] * W[j,k]   (A = attention output)
//   256 threads, 128x128x32 block tile, 8 waves (4x2), split-bf16 both sides,
//   software-pipelined global loads (staged in registers across compute).
// =====================================================================
__launch_bounds__(256, 1)
__global__ void proj_kernel(const float* __restrict__ Ag,
                            const float* __restrict__ Wg,
                            float*       __restrict__ Cg)
{
  const int bi   = blockIdx.x;        // 32 row tiles
  const int bj   = blockIdx.y;        // 16 col tiles
  const int tid  = threadIdx.x;
  const int lane = tid & 31;
  const int wv   = tid >> 5;
  const int l16  = lane & 15;
  const int lh   = (lane >> 4) & 1;
  const int mw   = wv & 3;            // wave row 0..3  (32 rows each)
  const int nw   = wv >> 2;           // wave col 0..1  (64 cols each)

  __shared__ __align__(16) unsigned short Ahi[128][40];
  __shared__ __align__(16) unsigned short Alo[128][40];
  __shared__ __align__(16) unsigned short Bhi[128][40];
  __shared__ __align__(16) unsigned short Blo[128][40];

  v8f c[2][4];
  #pragma unroll
  for (int ms = 0; ms < 2; ++ms)
    #pragma unroll
    for (int ns = 0; ns < 4; ++ns)
      #pragma unroll
      for (int e = 0; e < 8; ++e) c[ms][ns][e] = 0.0f;

  const float* ab = Ag + (size_t)bi * 128 * EMBED;
  const float* wb = Wg + (size_t)bj * 128 * EMBED;

  // cooperative-load geometry: 128 rows x 8 f32x4 over 256 threads -> 4 per thread
  int crow[4], cd4[4];
  #pragma unroll
  for (int i = 0; i < 4; ++i) {
    int idx = tid + i * 256;
    crow[i] = idx >> 3;
    cd4[i]  = (idx & 7) << 2;
  }

  // ---- prologue: stage K-step 0
  F4 areg[4], wreg[4];
  #pragma unroll
  for (int i = 0; i < 4; ++i) {
    areg[i] = *(const F4*)(ab + (size_t)crow[i] * EMBED + cd4[i]);
    wreg[i] = *(const F4*)(wb + (size_t)crow[i] * EMBED + cd4[i]);
  }

  for (int k0 = 0; k0 < EMBED; k0 += 32) {
    // ---- commit staged K-step to LDS
    __syncthreads();
    #pragma unroll
    for (int i = 0; i < 4; ++i) {
      F4 f = areg[i], g = wreg[i];
      int row = crow[i], d4 = cd4[i];
      float x0 = bfval(f.x), x1 = bfval(f.y), x2 = bfval(f.z), x3 = bfval(f.w);
      unsigned* ph = (unsigned*)&Ahi[row][d4];
      ph[0] = pk2(f.x, f.y); ph[1] = pk2(f.z, f.w);
      unsigned* pl = (unsigned*)&Alo[row][d4];
      pl[0] = pk2(f.x - x0, f.y - x1); pl[1] = pk2(f.z - x2, f.w - x3);
      float y0 = bfval(g.x), y1 = bfval(g.y), y2 = bfval(g.z), y3 = bfval(g.w);
      unsigned* qh = (unsigned*)&Bhi[row][d4];
      qh[0] = pk2(g.x, g.y); qh[1] = pk2(g.z, g.w);
      unsigned* ql = (unsigned*)&Blo[row][d4];
      ql[0] = pk2(g.x - y0, g.y - y1); ql[1] = pk2(g.z - y2, g.w - y3);
    }
    __syncthreads();

    // ---- issue next K-step's global loads; WMMAs below hide the latency
    if (k0 + 32 < EMBED) {
      #pragma unroll
      for (int i = 0; i < 4; ++i) {
        areg[i] = *(const F4*)(ab + (size_t)crow[i] * EMBED + k0 + 32 + cd4[i]);
        wreg[i] = *(const F4*)(wb + (size_t)crow[i] * EMBED + k0 + 32 + cd4[i]);
      }
    }

    v16bf afh[2], afl[2];
    #pragma unroll
    for (int ms = 0; ms < 2; ++ms) {
      Frag16 uh, ul;
      const unsigned short* rh = &Ahi[mw*32 + ms*16 + l16][0];
      const unsigned short* rl = &Alo[mw*32 + ms*16 + l16][0];
      uh.q[0] = *(const U4*)(rh + 16*lh); uh.q[1] = *(const U4*)(rh + 16*lh + 8);
      ul.q[0] = *(const U4*)(rl + 16*lh); ul.q[1] = *(const U4*)(rl + 16*lh + 8);
      afh[ms] = uh.v; afl[ms] = ul.v;
    }
    #pragma unroll
    for (int ns = 0; ns < 4; ++ns) {
      Frag16 uh, ul;
      const unsigned short* rh = &Bhi[nw*64 + ns*16 + l16][0];
      const unsigned short* rl = &Blo[nw*64 + ns*16 + l16][0];
      uh.q[0] = *(const U4*)(rh + 16*lh); uh.q[1] = *(const U4*)(rh + 16*lh + 8);
      ul.q[0] = *(const U4*)(rl + 16*lh); ul.q[1] = *(const U4*)(rl + 16*lh + 8);
      #pragma unroll
      for (int ms = 0; ms < 2; ++ms) {
        c[ms][ns] = __builtin_amdgcn_wmma_f32_16x16x32_bf16(false, afh[ms], false, uh.v, (short)0, c[ms][ns], false, false);
        c[ms][ns] = __builtin_amdgcn_wmma_f32_16x16x32_bf16(false, afh[ms], false, ul.v, (short)0, c[ms][ns], false, false);
        c[ms][ns] = __builtin_amdgcn_wmma_f32_16x16x32_bf16(false, afl[ms], false, uh.v, (short)0, c[ms][ns], false, false);
      }
    }
  }

  #pragma unroll
  for (int ms = 0; ms < 2; ++ms) {
    #pragma unroll
    for (int ns = 0; ns < 4; ++ns) {
      size_t i0 = (size_t)(bi*128 + mw*32 + ms*16 + 8*lh);
      int    j  = bj*128 + nw*64 + ns*16 + l16;
      #pragma unroll
      for (int r = 0; r < 8; ++r)
        Cg[(i0 + r) * EMBED + j] = c[ms][ns][r];
    }
  }
}

// =====================================================================
extern "C" void kernel_launch(void* const* d_in, const int* in_sizes, int n_in,
                              void* d_out, int out_size, void* d_ws, size_t ws_size,
                              hipStream_t stream) {
  (void)in_sizes; (void)n_in; (void)out_size; (void)ws_size;
  const float* V    = (const float*)d_in[0];
  const float* K    = (const float*)d_in[1];
  const float* Q    = (const float*)d_in[2];
  const int*   mask = (const int*)d_in[3];
  const float* W    = (const float*)d_in[4];
  float* out  = (float*)d_out;
  float* attn = (float*)d_ws;   // N*SEQ*EMBED fp32 = 32 MiB staging

  attn_kernel<<<dim3(NBATCH * HEADS * (SEQ / 128)), dim3(256), 0, stream>>>(V, K, Q, mask, attn);
  proj_kernel<<<dim3((NBATCH * SEQ) / 128, EMBED / 128), dim3(256), 0, stream>>>(attn, W, out);
}
